// VQVAE_42923903156709
// MI455X (gfx1250) — matrix operand order repaired and checked by reading, not compiled, not run on hardware
//
#include <hip/hip_runtime.h>
#include <hip/hip_bf16.h>
#include <stddef.h>
#include <stdint.h>

// ---------------------------------------------------------------------------
// VQ-VAE forward for MI455X (gfx1250, wave32, WMMA).
// bf16 activations/weights, f32 WMMA accumulation. Workload is HBM-bound
// (~2-3 GB of activation traffic at 23.3 TB/s); convs use 64px x 64ch tiles,
// double-buffered LDS staging, and 4 v_wmma_f32_16x16x32_bf16 per wave per
// K-step so compute and L2 traffic stay under the memory roofline.
// ---------------------------------------------------------------------------

typedef __bf16 bf16_t;
typedef __attribute__((ext_vector_type(16))) __bf16 v16bf;
typedef __attribute__((ext_vector_type(8)))  __bf16 v8bf;
typedef __attribute__((ext_vector_type(8)))  float  v8f;

#define WMMA_BF16(AF, BF, ACC) \
  __builtin_amdgcn_wmma_f32_16x16x32_bf16(false, (AF), false, (BF), (short)0, (ACC), false, false)

#define SHUF16(A0, A1) \
  __builtin_shufflevector((A0), (A1), 0,1,2,3,4,5,6,7,8,9,10,11,12,13,14,15)

// ---------------------------------------------------------------------------
// Implicit-GEMM conv, fused bias/skip/relu, optional stride-2 or fused 2x
// nearest upsample.  Block = 128 threads (4 waves).
// Tile: M=64 output pixels (one W-run) x N=64 output channels (16/wave).
// K loop over (kh,kw,cin/32), double-buffered LDS, 4 WMMA per wave per step
// (one B fragment reused across 4 A subtiles).
// Fragment layouts as used here (wave32; A/B in contiguous k order, not ISA table order):
//   A (16x32, MxK): lane L -> M=L%16; half  j   : K=(L/16)*16+j
//   B (32x16, KxN): lane L -> N=L%16; half  j   : K=(L/16)*16+j
//   D (16x16 f32) : lane L -> N=L%16; vgpr  r   : M=(L/16)*8+r
// ---------------------------------------------------------------------------
template<int KS, int STRIDE, int UPS, typename TIN, bool OUT_F32>
__global__ __launch_bounds__(128)
void conv_wmma_kernel(const TIN* __restrict__ in,
                      const bf16_t* __restrict__ wb,   // pre-converted bf16 weights (OIHW)
                      const float* __restrict__ bias,
                      const bf16_t* skip,              // may alias out (same-thread R->W)
                      void* out,
                      int Cin, int Hin, int Win,
                      int Cout, int Hout, int Wout, int relu)
{
  constexpr int PAD = (KS == 3) ? 1 : 0;
  const int t    = threadIdx.x;
  const int wave = t >> 5, lane = t & 31;
  const int hi   = lane >> 4, lo = lane & 15;

  const int oh    = blockIdx.y;
  const int slabs = (Cout + 63) >> 6;
  const int b     = blockIdx.z / slabs;
  const int slab  = blockIdx.z % slabs;
  const int ow0   = blockIdx.x * 64;

  const int csteps = (Cin + 31) >> 5;
  const int S      = KS * KS * csteps;
  const int Hv = Hin << UPS, Wv = Win << UPS;

  __shared__ alignas(64) bf16_t lA[2][64 * 32];   // [m][k]
  __shared__ alignas(64) bf16_t lB[2][64 * 32];   // [n][k] (linear in e = n*32+k)

  bf16_t ra[16];   // staged A elements (this thread's 16 of 64x32)
  bf16_t rb[16];   // staged B elements

  auto stage = [&](int s) {                       // global -> registers
    const int cs = s % csteps, s2 = s / csteps;
    const int kw = s2 % KS,    kh = s2 / KS;
    const int c0 = cs * 32;
    const int ihv = oh * STRIDE + kh - PAD;
    #pragma unroll
    for (int i = 0; i < 16; ++i) {
      const int e = t + i * 128;
      const int m = e & 63, k = e >> 6;
      const int c = c0 + k;
      const int iwv = (ow0 + m) * STRIDE + kw - PAD;
      float v = 0.0f;
      if (c < Cin && (unsigned)ihv < (unsigned)Hv && (unsigned)iwv < (unsigned)Wv)
        v = (float)in[(((size_t)b * Cin + c) * Hin + (ihv >> UPS)) * Win + (iwv >> UPS)];
      ra[i] = (bf16_t)v;
    }
    #pragma unroll
    for (int i = 0; i < 16; ++i) {
      const int e = t + i * 128;
      const int n = e >> 5, k = e & 31;
      const int nG = slab * 64 + n;
      const int c  = c0 + k;
      bf16_t v = (bf16_t)0.0f;
      if (nG < Cout && c < Cin)
        v = wb[((size_t)(nG * Cin + c) * KS + kh) * KS + kw];
      rb[i] = v;
    }
  };
  auto commit = [&](int buf) {                    // registers -> LDS
    #pragma unroll
    for (int i = 0; i < 16; ++i) {
      const int e = t + i * 128;
      lA[buf][(e & 63) * 32 + (e >> 6)] = ra[i];
      lB[buf][e] = rb[i];
    }
  };

  v8f acc[4] = {};

  stage(0);
  commit(0);
  for (int s = 0; s < S; ++s) {
    __syncthreads();                              // LDS buf (s&1) ready; prev reads done
    const int cur = s & 1;
    if (s + 1 < S) stage(s + 1);                  // overlap global loads with WMMAs
    const bf16_t* bp = lB[cur] + (wave * 16 + lo) * 32 + hi * 16;
    v16bf bfrag = *(const v16bf*)bp;
    #pragma unroll
    for (int sub = 0; sub < 4; ++sub) {
      const bf16_t* ap = lA[cur] + (sub * 16 + lo) * 32;
      v8bf a0 = *(const v8bf*)(ap + hi * 16);
      v8bf a1 = *(const v8bf*)(ap + hi * 16 + 8);
      v16bf afrag = SHUF16(a0, a1);
      acc[sub] = WMMA_BF16(afrag, bfrag, acc[sub]);
    }
    if (s + 1 < S) commit((s + 1) & 1);           // write the *other* buffer
  }

  // ---- epilogue: bias + skip + relu; 8 contiguous pixels per lane ----
  const int n = slab * 64 + wave * 16 + lo;
  if (n < Cout) {
    const float bv = bias ? bias[n] : 0.0f;
    #pragma unroll
    for (int sub = 0; sub < 4; ++sub) {
      const size_t obase = (((size_t)b * Cout + n) * Hout + oh) * Wout
                         + ow0 + sub * 16 + hi * 8;
      #pragma unroll
      for (int r = 0; r < 8; ++r) {
        float v = acc[sub][r] + bv;
        if (skip) v += (float)skip[obase + r];
        if (relu && v < 0.0f) v = 0.0f;
        if (OUT_F32) ((float*)out)[obase + r]  = v;
        else         ((bf16_t*)out)[obase + r] = (bf16_t)v;
      }
    }
  }
}

// ---------------------------------------------------------------------------
__global__ void cvt_bf16_kernel(const float* __restrict__ s, bf16_t* __restrict__ d, int n)
{
  const int i = blockIdx.x * blockDim.x + threadIdx.x;
  if (i < n) d[i] = (bf16_t)s[i];
}

__global__ void cb_norm_kernel(const float* __restrict__ cb, float* __restrict__ cnorm)
{
  const int n = blockIdx.x * blockDim.x + threadIdx.x;
  if (n >= 1024) return;
  float s = 0.0f;
  for (int k = 0; k < 128; ++k) { const float v = cb[(size_t)n * 128 + k]; s += v * v; }
  cnorm[n] = s;
}

__global__ void zero1_kernel(float* p) { if (threadIdx.x == 0 && blockIdx.x == 0) *p = 0.0f; }

// ---------------------------------------------------------------------------
// VQ argmin. BCHW->(-1,128) view makes each flat "vector" a contiguous
// 128-half run of z; ||z||^2 is row-constant so argmin over cnorm - 2*z.c.
// Block = 128 threads: 16 rows x 1024 codes (16 N-tiles of 64), K=128.
// ---------------------------------------------------------------------------
__global__ __launch_bounds__(128)
void vq_argmin_kernel(const bf16_t* __restrict__ z, const bf16_t* __restrict__ cbb,
                      const float* __restrict__ cnorm, int* __restrict__ idx)
{
  const int t = threadIdx.x, wave = t >> 5, lane = t & 31;
  const int hi = lane >> 4, lo = lane & 15;
  const int rowbase = blockIdx.x * 16;

  __shared__ alignas(64) bf16_t lZ[16 * 128];
  __shared__ alignas(64) bf16_t lC[64 * 128];
  __shared__ float rd[16][64];
  __shared__ int   ri[16][64];

  const bf16_t* zsrc = z + (size_t)rowbase * 128;
  #pragma unroll
  for (int i = 0; i < 16; ++i) lZ[t + i * 128] = zsrc[t + i * 128];

  float bestd[8]; int besti[8];
  #pragma unroll
  for (int r = 0; r < 8; ++r) { bestd[r] = 3.4e38f; besti[r] = 0; }

  for (int nt = 0; nt < 16; ++nt) {
    __syncthreads();
    const bf16_t* csrc = cbb + (size_t)nt * 8192;
    #pragma unroll
    for (int i = 0; i < 64; ++i) lC[t + i * 128] = csrc[t + i * 128];
    __syncthreads();
    v8f acc = {};
    #pragma unroll
    for (int kc = 0; kc < 4; ++kc) {
      const bf16_t* ap = lZ + lo * 128 + kc * 32;
      v8bf a0 = *(const v8bf*)(ap + hi * 16);
      v8bf a1 = *(const v8bf*)(ap + hi * 16 + 8);
      v16bf afrag = SHUF16(a0, a1);
      const bf16_t* bp = lC + (wave * 16 + lo) * 128 + kc * 32 + hi * 16;
      v16bf bfrag = *(const v16bf*)bp;
      acc = WMMA_BF16(afrag, bfrag, acc);
    }
    const int n = nt * 64 + wave * 16 + lo;
    const float cn = cnorm[n];
    #pragma unroll
    for (int r = 0; r < 8; ++r) {
      const float d = cn - 2.0f * acc[r];
      if (d < bestd[r]) { bestd[r] = d; besti[r] = n; }
    }
  }
  const int e = wave * 16 + lo;
  #pragma unroll
  for (int r = 0; r < 8; ++r) { const int m = hi * 8 + r; rd[m][e] = bestd[r]; ri[m][e] = besti[r]; }
  __syncthreads();
  if (t < 16) {
    float bd = rd[t][0]; int bi = ri[t][0];
    for (int j = 1; j < 64; ++j) if (rd[t][j] < bd) { bd = rd[t][j]; bi = ri[t][j]; }
    idx[rowbase + t] = bi;
  }
}

// ---------------------------------------------------------------------------
// q = codebook[idx] (bf16) and loss += 1.25 * mean((q-z)^2).
// ---------------------------------------------------------------------------
__global__ __launch_bounds__(128)
void vq_gather_loss_kernel(const int* __restrict__ idx, const float* __restrict__ cb,
                           const bf16_t* __restrict__ z, bf16_t* __restrict__ q,
                           float* loss)
{
  const int t = threadIdx.x;
  const int rowbase = blockIdx.x * 16;
  float local = 0.0f;
  for (int j = 0; j < 16; ++j) {
    const int r = rowbase + j;
    const int i = idx[r];
    const float cv = cb[(size_t)i * 128 + t];
    const bf16_t qv = (bf16_t)cv;
    const size_t off = (size_t)r * 128 + t;
    q[off] = qv;
    const float diff = (float)qv - (float)z[off];
    local += diff * diff;
  }
  __shared__ float red[128];
  red[t] = local; __syncthreads();
  #pragma unroll
  for (int s = 64; s > 0; s >>= 1) { if (t < s) red[t] += red[t + s]; __syncthreads(); }
  if (t == 0) atomicAdd(loss, red[0] * (1.25f / (131072.0f * 128.0f)));
}

// ---------------------------------------------------------------------------
extern "C" void kernel_launch(void* const* d_in, const int* in_sizes, int n_in,
                              void* d_out, int out_size, void* d_ws, size_t ws_size,
                              hipStream_t stream) {
  (void)in_sizes; (void)n_in; (void)out_size; (void)ws_size;

  const float* x        = (const float*)d_in[0];
  const float* enc_in_w = (const float*)d_in[1];  const float* enc_in_b = (const float*)d_in[2];
  const float* rb0_w1   = (const float*)d_in[3];  const float* rb0_b1   = (const float*)d_in[4];
  const float* rb0_w2   = (const float*)d_in[5];  const float* rb0_b2   = (const float*)d_in[6];
  const float* rb1_w1   = (const float*)d_in[7];  const float* rb1_b1   = (const float*)d_in[8];
  const float* rb1_w2   = (const float*)d_in[9];  const float* rb1_b2   = (const float*)d_in[10];
  const float* rb1_ws   = (const float*)d_in[11]; const float* rb1_bs   = (const float*)d_in[12];
  const float* down_w   = (const float*)d_in[13]; const float* down_b   = (const float*)d_in[14];
  const float* cb       = (const float*)d_in[15];
  const float* drb1_w1  = (const float*)d_in[16]; const float* drb1_b1  = (const float*)d_in[17];
  const float* drb1_w2  = (const float*)d_in[18]; const float* drb1_b2  = (const float*)d_in[19];
  const float* up_w     = (const float*)d_in[20]; const float* up_b     = (const float*)d_in[21];
  const float* drb0_w1  = (const float*)d_in[22]; const float* drb0_b1  = (const float*)d_in[23];
  const float* drb0_w2  = (const float*)d_in[24]; const float* drb0_b2  = (const float*)d_in[25];
  const float* out_w    = (const float*)d_in[26]; const float* out_b    = (const float*)d_in[27];

  // Workspace carving: bf16 activations (3x 134MB ping-pong), z/q, idx/cnorm,
  // then a ~2MB region of pre-converted bf16 weights + codebook.
  char* wsb = (char*)d_ws;
  bf16_t* R0  = (bf16_t*)(wsb + 0);
  bf16_t* R1  = (bf16_t*)(wsb + 134217728);
  bf16_t* R2  = (bf16_t*)(wsb + 268435456);
  bf16_t* Z   = (bf16_t*)(wsb + 402653184);
  bf16_t* Q   = (bf16_t*)(wsb + 436207616);
  int*    IDX = (int*)  (wsb + 469762048);
  float*  CN  = (float*)(wsb + 470286336);
  bf16_t* WB  = (bf16_t*)(wsb + 470290432);

  size_t o = 0;
  bf16_t* wb_enc_in = WB + o; o += 64 * 3 * 9;      // 1728
  bf16_t* wb_rb0_w1 = WB + o; o += 64 * 64 * 9;     // 36864
  bf16_t* wb_rb0_w2 = WB + o; o += 64 * 64 * 9;
  bf16_t* wb_rb1_w1 = WB + o; o += 128 * 64 * 9;    // 73728
  bf16_t* wb_rb1_w2 = WB + o; o += 128 * 128 * 9;   // 147456
  bf16_t* wb_rb1_ws = WB + o; o += 128 * 64;        // 8192
  bf16_t* wb_down   = WB + o; o += 128 * 128 * 9;
  bf16_t* wb_drb1_1 = WB + o; o += 128 * 128 * 9;
  bf16_t* wb_drb1_2 = WB + o; o += 128 * 128 * 9;
  bf16_t* wb_up     = WB + o; o += 64 * 128 * 9;
  bf16_t* wb_drb0_1 = WB + o; o += 64 * 64 * 9;
  bf16_t* wb_drb0_2 = WB + o; o += 64 * 64 * 9;
  bf16_t* wb_out    = WB + o; o += 3 * 64 * 9;
  bf16_t* wb_cb     = WB + o; o += 1024 * 128;

  float* recon = (float*)d_out;
  float* lossp = recon + (size_t)8 * 3 * 256 * 256;

  const dim3 blk(128);
  const dim3 g256_64 (4, 256, 8);     // 256x256 out, Cout<=64  (1 slab)
  const dim3 g256_128(4, 256, 16);    // 256x256 out, Cout=128  (2 slabs)
  const dim3 g128_128(2, 128, 16);    // 128x128 out, Cout=128  (2 slabs)

  // ---- one-time conversions + quantizer prep + loss init ----
  auto cvt = [&](const float* s, bf16_t* d, int n) {
    cvt_bf16_kernel<<<(n + 255) / 256, 256, 0, stream>>>(s, d, n);
  };
  cvt(enc_in_w, wb_enc_in, 64 * 3 * 9);
  cvt(rb0_w1,   wb_rb0_w1, 64 * 64 * 9);
  cvt(rb0_w2,   wb_rb0_w2, 64 * 64 * 9);
  cvt(rb1_w1,   wb_rb1_w1, 128 * 64 * 9);
  cvt(rb1_w2,   wb_rb1_w2, 128 * 128 * 9);
  cvt(rb1_ws,   wb_rb1_ws, 128 * 64);
  cvt(down_w,   wb_down,   128 * 128 * 9);
  cvt(drb1_w1,  wb_drb1_1, 128 * 128 * 9);
  cvt(drb1_w2,  wb_drb1_2, 128 * 128 * 9);
  cvt(up_w,     wb_up,     64 * 128 * 9);
  cvt(drb0_w1,  wb_drb0_1, 64 * 64 * 9);
  cvt(drb0_w2,  wb_drb0_2, 64 * 64 * 9);
  cvt(out_w,    wb_out,    3 * 64 * 9);
  cvt(cb,       wb_cb,     1024 * 128);
  cb_norm_kernel<<<8, 128, 0, stream>>>(cb, CN);
  zero1_kernel<<<1, 1, 0, stream>>>(lossp);

  // ---- encoder ----
  conv_wmma_kernel<3,1,0,float, false><<<g256_64, blk, 0, stream>>>(
      x, wb_enc_in, enc_in_b, nullptr, R0, 3, 256, 256, 64, 256, 256, 0);
  conv_wmma_kernel<3,1,0,bf16_t,false><<<g256_64, blk, 0, stream>>>(
      R0, wb_rb0_w1, rb0_b1, nullptr, R1, 64, 256, 256, 64, 256, 256, 1);
  conv_wmma_kernel<3,1,0,bf16_t,false><<<g256_64, blk, 0, stream>>>(
      R1, wb_rb0_w2, rb0_b2, R0, R0, 64, 256, 256, 64, 256, 256, 1);      // skip==out OK
  conv_wmma_kernel<1,1,0,bf16_t,false><<<g256_128, blk, 0, stream>>>(
      R0, wb_rb1_ws, rb1_bs, nullptr, R1, 64, 256, 256, 128, 256, 256, 0);
  conv_wmma_kernel<3,1,0,bf16_t,false><<<g256_128, blk, 0, stream>>>(
      R0, wb_rb1_w1, rb1_b1, nullptr, R2, 64, 256, 256, 128, 256, 256, 1);
  conv_wmma_kernel<3,1,0,bf16_t,false><<<g256_128, blk, 0, stream>>>(
      R2, wb_rb1_w2, rb1_b2, R1, R1, 128, 256, 256, 128, 256, 256, 1);
  conv_wmma_kernel<3,2,0,bf16_t,false><<<g128_128, blk, 0, stream>>>(
      R1, wb_down, down_b, nullptr, Z, 128, 256, 256, 128, 128, 128, 1);  // stride-2 + relu

  // ---- vector quantizer ----
  vq_argmin_kernel<<<8192, blk, 0, stream>>>(Z, wb_cb, CN, IDX);
  vq_gather_loss_kernel<<<8192, blk, 0, stream>>>(IDX, cb, Z, Q, lossp);

  // ---- decoder ----
  conv_wmma_kernel<3,1,0,bf16_t,false><<<g128_128, blk, 0, stream>>>(
      Q, wb_drb1_1, drb1_b1, nullptr, R0, 128, 128, 128, 128, 128, 128, 1);
  conv_wmma_kernel<3,1,0,bf16_t,false><<<g128_128, blk, 0, stream>>>(
      R0, wb_drb1_2, drb1_b2, Q, Q, 128, 128, 128, 128, 128, 128, 1);     // skip==out OK
  conv_wmma_kernel<3,1,1,bf16_t,false><<<g256_64, blk, 0, stream>>>(
      Q, wb_up, up_b, nullptr, R1, 128, 128, 128, 64, 256, 256, 1);       // fused 2x NN upsample
  conv_wmma_kernel<3,1,0,bf16_t,false><<<g256_64, blk, 0, stream>>>(
      R1, wb_drb0_1, drb0_b1, nullptr, R0, 64, 256, 256, 64, 256, 256, 1);
  conv_wmma_kernel<3,1,0,bf16_t,false><<<g256_64, blk, 0, stream>>>(
      R0, wb_drb0_2, drb0_b2, R1, R1, 64, 256, 256, 64, 256, 256, 1);
  conv_wmma_kernel<3,1,0,bf16_t,true ><<<g256_64, blk, 0, stream>>>(
      R1, wb_out, out_b, nullptr, recon, 64, 256, 256, 3, 256, 256, 0);   // f32 recon out
}
